// MMP_Balance2_73229192397366
// MI455X (gfx1250) — matrix operand order repaired and checked by reading, not compiled, not run on hardware
//
#include <hip/hip_runtime.h>
#include <hip/hip_bf16.h>
#include <math.h>

// ---------------------------------------------------------------------------
// MMP_Balance2 loss, fused for MI455X (gfx1250, wave32, WMMA bf16 16x16x32)
// B=512, M=2, D=512, C=100000
// ---------------------------------------------------------------------------

typedef __attribute__((ext_vector_type(16))) __bf16 v16bf;
typedef __attribute__((ext_vector_type(8)))  __bf16 v8bf;
typedef __attribute__((ext_vector_type(8)))  float  v8f;

#define B_   512
#define D_   512
#define C_   100000
#define EPSF 1e-12f
#define NT   32                 // proxies per block in the big fused GEMM
#define KSTEPS (D_ / 32)        // 16 k-steps of 32 for bf16 WMMA

union ABf { v16bf v; v8bf h[2]; };

// A-fragment: lane<16 holds row M=l15, K = kb..kb+7 and kb+16..kb+23
// (caller folds hi16*8 into the base pointer; kb = ks*32)
__device__ __forceinline__ v16bf make_afrag(const __bf16* rowPtr, int kb) {
    ABf u;
    u.h[0] = *(const v8bf*)(rowPtr + kb);
    u.h[1] = *(const v8bf*)(rowPtr + kb + 16);
    return u.v;
}
// 16 contiguous bf16 from p (unused: B-fragments are loaded with make_afrag,
// hi16*8 folded into base pointer, same K order as A)
__device__ __forceinline__ v16bf load_contig16(const __bf16* p) {
    ABf u;
    u.h[0] = *(const v8bf*)(p);
    u.h[1] = *(const v8bf*)(p + 8);
    return u.v;
}

__device__ __forceinline__ float bsum256(float* red, int t, float v) {
    red[t] = v; __syncthreads();
    #pragma unroll
    for (int s = 128; s > 0; s >>= 1) { if (t < s) red[t] += red[t + s]; __syncthreads(); }
    float r = red[0]; __syncthreads();
    return r;
}
__device__ __forceinline__ float bmax256(float* red, int t, float v) {
    red[t] = v; __syncthreads();
    #pragma unroll
    for (int s = 128; s > 0; s >>= 1) { if (t < s) red[t] = fmaxf(red[t], red[t + s]); __syncthreads(); }
    float r = red[0]; __syncthreads();
    return r;
}

// ---------------------------------------------------------------------------
// Kernel 1: per-row prep. pos/anchor -> bf16, anchor_n = anchor/sqrt(ss+EPS).
// Also zeroes L2full (graph-safe re-init every launch).
// ---------------------------------------------------------------------------
__global__ __launch_bounds__(256)
void prep_kernel(const float* __restrict__ X,
                 __bf16* __restrict__ posb, __bf16* __restrict__ anchb,
                 __bf16* __restrict__ anchnb, float* __restrict__ L2full)
{
    __shared__ float red[256];
    const int i = blockIdx.x, t = threadIdx.x;
    const float* pos = X + (size_t)i * 2 * D_;
    const float* anc = pos + D_;
    const float a0 = anc[t], a1 = anc[t + 256];
    const float ss = bsum256(red, t, a0 * a0 + a1 * a1);
    const float rinv = 1.0f / sqrtf(ss + EPSF);
    const size_t base = (size_t)i * D_;
    posb[base + t]         = (__bf16)pos[t];
    posb[base + t + 256]   = (__bf16)pos[t + 256];
    anchb[base + t]        = (__bf16)a0;
    anchb[base + t + 256]  = (__bf16)a1;
    anchnb[base + t]       = (__bf16)(a0 * rinv);
    anchnb[base + t + 256] = (__bf16)(a1 * rinv);
    if (t == 0) L2full[i] = 0.0f;
}

// ---------------------------------------------------------------------------
// Kernel 2: gather normalized proxies at labels T: PTnb[k] = proxies[T[k]]/||.||
// ---------------------------------------------------------------------------
__global__ __launch_bounds__(256)
void gather_kernel(const float* __restrict__ proxies, const int* __restrict__ T,
                   __bf16* __restrict__ PTnb)
{
    __shared__ float red[256];
    const int k = blockIdx.x, t = threadIdx.x;
    const int j = T[k];
    const float* pr = proxies + (size_t)j * D_;
    const float a0 = pr[t], a1 = pr[t + 256];
    const float ss = bsum256(red, t, a0 * a0 + a1 * a1);
    const float rinv = 1.0f / fmaxf(sqrtf(ss), EPSF);
    PTnb[(size_t)k * D_ + t]       = (__bf16)(a0 * rinv);
    PTnb[(size_t)k * D_ + t + 256] = (__bf16)(a1 * rinv);
}

// ---------------------------------------------------------------------------
// Kernel 3: heavy fused GEMM with software-pipelined K loop.
// For each proxy j: S[i,j] = (pos_i . proxies_j) * inv_norm_j ;
// L2full[i] += exp(S[i,j]*w - b), summed over all j (T columns subtracted later).
// Block = 256 thr (8 waves), NT=32 proxies, full M=512, K=512. grid = 3125.
// Double-buffered fragments: loads for k-step ks+1 are issued before the 8
// WMMAs of ks consume the other buffer -> loads overlap matrix math (no WAR).
// ---------------------------------------------------------------------------
__global__ __launch_bounds__(256)
void big_gemm_exp_kernel(const float* __restrict__ proxies,
                         const __bf16* __restrict__ posb,
                         const float* __restrict__ wp, const float* __restrict__ bp,
                         float* __restrict__ L2full)
{
    __shared__ __bf16 sProx[NT][D_ + 8];   // +8 bf16 pad -> conflict-free b128 reads
    __shared__ float sInv[NT];
    __shared__ float sSumsq[NT];
    __shared__ float sRow[B_];

    const int t = threadIdx.x;
    const float w  = wp[0];
    const float bb = bp[0];

    for (int i = t; i < B_; i += 256) sRow[i] = 0.0f;
    if (t < NT) sSumsq[t] = 0.0f;
    __syncthreads();

    // Stage 32x512 fp32 proxy tile -> bf16 LDS, fp32 row sum-of-squares on the fly.
    {
        const int row = t >> 3;                 // 0..31
        const int seg = t & 7;                  // 8 threads/row, 64 elems each
        const float* gp = proxies + (size_t)(blockIdx.x * NT + row) * D_ + seg * 64;
        float ss = 0.0f;
        #pragma unroll 8
        for (int e = 0; e < 64; ++e) {
            const float v = gp[e];
            ss += v * v;
            sProx[row][seg * 64 + e] = (__bf16)v;
        }
        atomicAdd(&sSumsq[row], ss);
    }
    __syncthreads();
    if (t < NT) sInv[t] = 1.0f / fmaxf(sqrtf(sSumsq[t]), EPSF);
    __syncthreads();

    const int wv    = t >> 5;
    const int lane  = t & 31;
    const int l15   = lane & 15;
    const int hi16  = (lane >> 4) & 1;
    const int ntl   = (wv & 1) * 16;          // local proxy-column base
    const int mbase = (wv >> 1) * 128;        // row base (8 M-tiles of 16)

    v8f acc[8];
    const v8f vz = {0.f, 0.f, 0.f, 0.f, 0.f, 0.f, 0.f, 0.f};
    #pragma unroll
    for (int tt = 0; tt < 8; ++tt) acc[tt] = vz;

    // Per-lane fragment bases (hi16 selection folded in once).
    const __bf16* abase = posb + (size_t)(mbase + l15) * D_ + hi16 * 8;
    const __bf16* bbase = &sProx[ntl + l15][hi16 * 8];

    v16bf af[2][8];
    v16bf bf[2];

    // Prologue: stage 0.
    #pragma unroll
    for (int tt = 0; tt < 8; ++tt) af[0][tt] = make_afrag(abase + tt * 16 * D_, 0);
    bf[0] = make_afrag(bbase, 0);

    #pragma unroll
    for (int ks = 0; ks < KSTEPS; ++ks) {
        const int c = ks & 1;
        const int n = c ^ 1;
        if (ks + 1 < KSTEPS) {
            const int kb = (ks + 1) * 32;
            #pragma unroll
            for (int tt = 0; tt < 8; ++tt)
                af[n][tt] = make_afrag(abase + tt * 16 * D_, kb);
            bf[n] = make_afrag(bbase, kb);
        }
        #pragma unroll
        for (int tt = 0; tt < 8; ++tt)
            acc[tt] = __builtin_amdgcn_wmma_f32_16x16x32_bf16(
                false, af[c][tt], false, bf[c], (short)0, acc[tt], false, false);
    }

    // Epilogue: scale by inv_norm_j, exp(w*S-b), reduce over the 16 j-lanes,
    // then one conflict-free LDS atomic per row.
    const float inv = sInv[ntl + l15];
    #pragma unroll
    for (int tt = 0; tt < 8; ++tt) {
        const int rbase = mbase + tt * 16 + hi16 * 8;
        #pragma unroll
        for (int r = 0; r < 8; ++r) {
            float e = __expf(acc[tt][r] * inv * w - bb);
            e += __shfl_xor(e, 1, 16);
            e += __shfl_xor(e, 2, 16);
            e += __shfl_xor(e, 4, 16);
            e += __shfl_xor(e, 8, 16);
            if (l15 == 0) atomicAdd(&sRow[rbase + r], e);
        }
    }
    __syncthreads();
    for (int i = t; i < B_; i += 256) atomicAdd(&L2full[i], sRow[i]);
}

// ---------------------------------------------------------------------------
// Kernel 4: 512x512x512 GEMM, out = A @ Bm^T (row-major [512][512] bf16),
// double-buffered K loop, one 16x16 tile per wave. grid = (32, 4), 256 thr.
// ---------------------------------------------------------------------------
__global__ __launch_bounds__(256)
void small_gemm_kernel(const __bf16* __restrict__ A,
                       const __bf16* __restrict__ Bm,
                       float* __restrict__ out)
{
    const int t = threadIdx.x;
    const int wv = t >> 5, lane = t & 31;
    const int l15 = lane & 15, hi16 = (lane >> 4) & 1;
    const int mt = blockIdx.y * 8 + wv;   // 0..31
    const int nt = blockIdx.x;            // 0..31

    v8f acc = {0.f, 0.f, 0.f, 0.f, 0.f, 0.f, 0.f, 0.f};
    const __bf16* arow = A  + (size_t)(mt * 16 + l15) * D_ + hi16 * 8;
    const __bf16* brow = Bm + (size_t)(nt * 16 + l15) * D_ + hi16 * 8;

    v16bf af[2], bf[2];
    af[0] = make_afrag(arow, 0);
    bf[0] = make_afrag(brow, 0);
    #pragma unroll
    for (int ks = 0; ks < KSTEPS; ++ks) {
        const int c = ks & 1, n = c ^ 1;
        if (ks + 1 < KSTEPS) {
            const int kb = (ks + 1) * 32;
            af[n] = make_afrag(arow, kb);
            bf[n] = make_afrag(brow, kb);
        }
        acc = __builtin_amdgcn_wmma_f32_16x16x32_bf16(
            false, af[c], false, bf[c], (short)0, acc, false, false);
    }
    const int j  = nt * 16 + l15;
    const int ib = mt * 16 + hi16 * 8;
    #pragma unroll
    for (int r = 0; r < 8; ++r) out[(size_t)(ib + r) * B_ + j] = acc[r];
}

// ---------------------------------------------------------------------------
// Kernel 5: per-row loss pieces.
//  rowL2[i]  = log( L2full[i] - sum_k exp(w*A2[i,k]-b) + 1 )
//  rowL3[i]  = log( sum_k exp(w*Z[i,k]-b) - exp(w*Z[i,i]-b) + 1 )
//  rowReg[i] = logsumexp_k(w*A3[i,k]-b) - (w*A3[i,i]-b)
//  expNeg[i] = exp(-w*Z[i,i] + b)
// ---------------------------------------------------------------------------
__global__ __launch_bounds__(256)
void rowloss_kernel(const float* __restrict__ Z, const float* __restrict__ A2,
                    const float* __restrict__ A3, const float* __restrict__ L2full,
                    const float* __restrict__ wp, const float* __restrict__ bp,
                    float* __restrict__ rowL2, float* __restrict__ rowL3,
                    float* __restrict__ rowReg, float* __restrict__ expNeg)
{
    __shared__ float red[256];
    const int i = blockIdx.x, t = threadIdx.x;
    const float w = wp[0], bb = bp[0];
    const float* zr  = Z  + (size_t)i * B_;
    const float* a2r = A2 + (size_t)i * B_;
    const float* a3r = A3 + (size_t)i * B_;

    const float mloc = fmaxf(a3r[t] * w - bb, a3r[t + 256] * w - bb);
    const float mx = bmax256(red, t, mloc);

    float sZ = 0.f, s2 = 0.f, s3 = 0.f;
    for (int k = t; k < B_; k += 256) {
        sZ += __expf(zr[k]  * w - bb);
        s2 += __expf(a2r[k] * w - bb);
        s3 += __expf(a3r[k] * w - bb - mx);
    }
    const float sumZ  = bsum256(red, t, sZ);
    const float sumA2 = bsum256(red, t, s2);
    const float sumC  = bsum256(red, t, s3);

    if (t == 0) {
        const float zii = zr[i];
        rowL2[i]  = logf(L2full[i] - sumA2 + 1.0f);
        rowL3[i]  = logf(sumZ - __expf(zii * w - bb) + 1.0f);
        rowReg[i] = (mx + logf(sumC)) - (a3r[i] * w - bb);
        expNeg[i] = __expf(-zii * w + bb);
    }
}

// ---------------------------------------------------------------------------
// Kernel 6: final scalar. loss = log(sum expNeg + 1) + mean(rowL2) + mean(rowL3)
//                              + 0.5 * mean(rowReg)
// ---------------------------------------------------------------------------
__global__ __launch_bounds__(256)
void final_kernel(const float* __restrict__ rowL2, const float* __restrict__ rowL3,
                  const float* __restrict__ rowReg, const float* __restrict__ expNeg,
                  float* __restrict__ out)
{
    __shared__ float red[256];
    const int t = threadIdx.x;
    const float sE = bsum256(red, t, expNeg[t] + expNeg[t + 256]);
    const float s2 = bsum256(red, t, rowL2[t]  + rowL2[t + 256]);
    const float s3 = bsum256(red, t, rowL3[t]  + rowL3[t + 256]);
    const float sR = bsum256(red, t, rowReg[t] + rowReg[t + 256]);
    if (t == 0)
        out[0] = logf(sE + 1.0f) + s2 * (1.0f / 512.0f) + s3 * (1.0f / 512.0f)
               + 0.5f * (sR * (1.0f / 512.0f));
}

// ---------------------------------------------------------------------------
extern "C" void kernel_launch(void* const* d_in, const int* in_sizes, int n_in,
                              void* d_out, int out_size, void* d_ws, size_t ws_size,
                              hipStream_t stream)
{
    (void)in_sizes; (void)n_in; (void)out_size; (void)ws_size;
    const float* X       = (const float*)d_in[0];   // (512,2,512) f32
    const float* proxies = (const float*)d_in[1];   // (100000,512) f32
    const float* wp      = (const float*)d_in[2];   // scalar f32
    const float* bp      = (const float*)d_in[3];   // scalar f32
    const int*   T       = (const int*)d_in[4];     // (512,) i32
    float* out = (float*)d_out;

    char* ws = (char*)d_ws;
    size_t off = 0;
    auto alloc = [&](size_t bytes) -> void* {
        void* p = (void*)(ws + off);
        off += (bytes + 255) & ~(size_t)255;
        return p;
    };
    __bf16* posb   = (__bf16*)alloc((size_t)B_ * D_ * 2);
    __bf16* anchb  = (__bf16*)alloc((size_t)B_ * D_ * 2);
    __bf16* anchnb = (__bf16*)alloc((size_t)B_ * D_ * 2);
    __bf16* PTnb   = (__bf16*)alloc((size_t)B_ * D_ * 2);
    float*  Zmat   = (float*)alloc((size_t)B_ * B_ * 4);
    float*  A2     = (float*)alloc((size_t)B_ * B_ * 4);
    float*  A3     = (float*)alloc((size_t)B_ * B_ * 4);
    float*  L2full = (float*)alloc((size_t)B_ * 4);
    float*  rowL2  = (float*)alloc((size_t)B_ * 4);
    float*  rowL3  = (float*)alloc((size_t)B_ * 4);
    float*  rowReg = (float*)alloc((size_t)B_ * 4);
    float*  expNeg = (float*)alloc((size_t)B_ * 4);

    prep_kernel<<<B_, 256, 0, stream>>>(X, posb, anchb, anchnb, L2full);
    gather_kernel<<<B_, 256, 0, stream>>>(proxies, T, PTnb);
    big_gemm_exp_kernel<<<C_ / NT, 256, 0, stream>>>(proxies, posb, wp, bp, L2full);
    dim3 g4(32, 4);
    small_gemm_kernel<<<g4, 256, 0, stream>>>(posb,  anchnb, Zmat);  // Z  = pos . anchor_n^T
    small_gemm_kernel<<<g4, 256, 0, stream>>>(posb,  PTnb,   A2);    // A2 = pos . P[T]n^T
    small_gemm_kernel<<<g4, 256, 0, stream>>>(anchb, PTnb,   A3);    // A3 = anchor . P[T]n^T
    rowloss_kernel<<<B_, 256, 0, stream>>>(Zmat, A2, A3, L2full, wp, bp,
                                           rowL2, rowL3, rowReg, expNeg);
    final_kernel<<<1, 256, 0, stream>>>(rowL2, rowL3, rowReg, expNeg, out);
}
